// PrimIntentEmbeddingVQ_87883620811207
// MI455X (gfx1250) — matrix operand order repaired and checked by reading, not compiled, not run on hardware
//
#include <hip/hip_runtime.h>
#include <hip/hip_bf16.h>
#include <stdint.h>

// ---------------------------------------------------------------------------
// PrimIntentEmbeddingVQ on MI455X (gfx1250):
//   x = concat(skills, lang)            [2048,1024] fp32 -> bf16
//   h1 = relu(x @ W1 + b1)              WMMA bf16, f32 acc
//   h2 = relu(h1 @ W2 + b2)             WMMA bf16, f32 acc
//   u  = h2 @ W3 + b3                   WMMA bf16, f32 acc -> f32 out + bf16 copy
//   dot = u @ codebook^T                WMMA bf16, f32 acc
//   idx = argmin_j (||c_j||^2 - 2*dot)  (== argmin L2 distance)
//   q  = codebook[idx]                  (fp32 gather)
// GEMM inner loop is software-pipelined (double-buffered fragments) so the
// global_load_b128 clauses for K-step k+32 overlap the 8-WMMA burst for k.
// ---------------------------------------------------------------------------

typedef __attribute__((ext_vector_type(16))) __bf16 v16bf;
typedef __attribute__((ext_vector_type(8)))  float  v8f;

struct B256 { uint4 lo, hi; };   // 32 bytes == one v16bf fragment per lane

static __device__ __forceinline__ unsigned short f32_to_bf16_rne(float f) {
  uint32_t u = __builtin_bit_cast(uint32_t, f);
  uint32_t r = u + 0x7FFFu + ((u >> 16) & 1u);
  return (unsigned short)(r >> 16);
}

// A 16x32 bf16 fragment (ISA 16-bit A layout): lane-half hi holds K chunks
// {k0+8*hi..+7} and {k0+16+8*hi..+7}.
static __device__ __forceinline__ v16bf load_a_frag(
    const unsigned short* __restrict__ rowp, int k0, int hi) {
  const unsigned short* p = rowp + k0 + hi * 8;
  B256 t;
  t.lo = *(const uint4*)(p);
  t.hi = *(const uint4*)(p + 16);        // +16 elements == +32 bytes
  return __builtin_bit_cast(v16bf, t);
}

// B 32x16 bf16 fragment from Bt rows (same K order as A): lane-half
// hi holds K chunks {k0+8*hi..+7} and {k0+16+8*hi..+7}.
static __device__ __forceinline__ v16bf load_b_frag(
    const unsigned short* __restrict__ rowp, int k0, int hi) {
  const unsigned short* p = rowp + k0 + hi * 8;
  B256 t;
  t.lo = *(const uint4*)(p);
  t.hi = *(const uint4*)(p + 16);
  return __builtin_bit_cast(v16bf, t);
}

// ---- conversion kernels ----------------------------------------------------

// Xbf[b, 0:512] = skills[b], Xbf[b, 512:1024] = lang[b]  (fp32 -> bf16)
__global__ void k_concat_cvt(const float* __restrict__ s,
                             const float* __restrict__ l,
                             unsigned short* __restrict__ x) {
  int i = blockIdx.x * blockDim.x + threadIdx.x;      // over 2048*1024
  int b = i >> 10, c = i & 1023;
  float v = (c < 512) ? s[b * 512 + c] : l[b * 512 + (c - 512)];
  x[i] = f32_to_bf16_rne(v);
}

// wt[n, k] = (bf16) w[k, n]   (w is [K, N] row-major)
__global__ void k_transpose_cvt(const float* __restrict__ w,
                                unsigned short* __restrict__ wt,
                                int K, int N) {
  int i = blockIdx.x * blockDim.x + threadIdx.x;      // over N*K
  int n = i / K, k = i - n * K;
  wt[i] = f32_to_bf16_rne(w[(size_t)k * N + n]);
}

__global__ void k_copy_cvt(const float* __restrict__ a,
                           unsigned short* __restrict__ o, int n) {
  int i = blockIdx.x * blockDim.x + threadIdx.x;
  if (i < n) o[i] = f32_to_bf16_rne(a[i]);
}

// cbn[j] = sum_k codebook[j,k]^2   (fp32, one block per code)
__global__ void k_cbnorm(const float* __restrict__ cb, float* __restrict__ cbn) {
  __shared__ float red[128];
  const float* row = cb + (size_t)blockIdx.x * 512;
  float s = 0.f;
  for (int c = threadIdx.x; c < 512; c += 128) { float v = row[c]; s += v * v; }
  red[threadIdx.x] = s;
  __syncthreads();
  for (int off = 64; off > 0; off >>= 1) {
    if (threadIdx.x < off) red[threadIdx.x] += red[threadIdx.x + off];
    __syncthreads();
  }
  if (threadIdx.x == 0) cbn[blockIdx.x] = red[0];
}

// ---- WMMA GEMM: C[M,N] = A[M,K] @ Bt[N,K]^T (+bias) (+relu) ---------------
// Workgroup: 256 threads = 8 waves, tile 128(M) x 128(N).
// Wave: 64(M) x 32(N) = 4x2 fragments of v_wmma_f32_16x16x32_bf16.
// K must be a multiple of 64 (true for all call sites here).
__global__ __launch_bounds__(256) void k_gemm_bf16(
    const unsigned short* __restrict__ A,    // [M,K] bf16 row-major
    const unsigned short* __restrict__ Bt,   // [N,K] bf16 row-major (== B^T)
    const float* __restrict__ bias,          // [N] or nullptr
    float* __restrict__ outF,                // [M,N] f32 or nullptr
    unsigned short* __restrict__ outB,       // [M,N] bf16 or nullptr
    int M, int N, int K, int relu) {
  const int lane = threadIdx.x & 31;
  const int wave = threadIdx.x >> 5;       // 0..7
  const int wm = wave >> 2;                // 0..1
  const int wn = wave & 3;                 // 0..3
  const int l  = lane & 15;
  const int hi = lane >> 4;                // lane-half selects K chunk

  const int mBase = blockIdx.y * 128 + wm * 64;
  const int nBase = blockIdx.x * 128 + wn * 32;

  const v8f zero = {0.f, 0.f, 0.f, 0.f, 0.f, 0.f, 0.f, 0.f};
  v8f acc[4][2];
#pragma unroll
  for (int mi = 0; mi < 4; ++mi)
#pragma unroll
    for (int ni = 0; ni < 2; ++ni) acc[mi][ni] = zero;

  // Per-lane row pointers (rows constant across the K loop).
  const unsigned short* arow[4];
#pragma unroll
  for (int mi = 0; mi < 4; ++mi)
    arow[mi] = A + (size_t)(mBase + mi * 16 + l) * K;
  const unsigned short* brow[2];
#pragma unroll
  for (int ni = 0; ni < 2; ++ni)
    brow[ni] = Bt + (size_t)(nBase + ni * 16 + l) * K;

  // Software pipeline: double-buffered fragments, 2 K-steps per iteration.
  v16bf a0[4], b0[2], a1[4], b1[2];
#pragma unroll
  for (int mi = 0; mi < 4; ++mi) a0[mi] = load_a_frag(arow[mi], 0, hi);
#pragma unroll
  for (int ni = 0; ni < 2; ++ni) b0[ni] = load_b_frag(brow[ni], 0, hi);

  for (int k0 = 0; k0 < K; k0 += 64) {
    // Prefetch the K tile two steps ahead (global_prefetch_b8).
    if (k0 + 64 < K) {
      __builtin_prefetch(arow[0] + k0 + 64, 0, 1);
      __builtin_prefetch(brow[0] + k0 + 64, 0, 1);
    }
    // Issue loads for K-step k0+32 (always in range: k0 <= K-64).
#pragma unroll
    for (int mi = 0; mi < 4; ++mi) a1[mi] = load_a_frag(arow[mi], k0 + 32, hi);
#pragma unroll
    for (int ni = 0; ni < 2; ++ni) b1[ni] = load_b_frag(brow[ni], k0 + 32, hi);
    // Compute on K-step k0 while the above loads are in flight.
#pragma unroll
    for (int mi = 0; mi < 4; ++mi)
#pragma unroll
      for (int ni = 0; ni < 2; ++ni)
        acc[mi][ni] = __builtin_amdgcn_wmma_f32_16x16x32_bf16(
            false, a0[mi], false, b0[ni], (short)0, acc[mi][ni], false, false);
    // Issue loads for K-step k0+64 (dummy reload of tile 0 on last iter —
    // keeps control flow uniform; WMMA requires EXEC all-ones).
    const int kn = (k0 + 64 < K) ? (k0 + 64) : 0;
#pragma unroll
    for (int mi = 0; mi < 4; ++mi) a0[mi] = load_a_frag(arow[mi], kn, hi);
#pragma unroll
    for (int ni = 0; ni < 2; ++ni) b0[ni] = load_b_frag(brow[ni], kn, hi);
    // Compute on K-step k0+32.
#pragma unroll
    for (int mi = 0; mi < 4; ++mi)
#pragma unroll
      for (int ni = 0; ni < 2; ++ni)
        acc[mi][ni] = __builtin_amdgcn_wmma_f32_16x16x32_bf16(
            false, a1[mi], false, b1[ni], (short)0, acc[mi][ni], false, false);
  }

  // Epilogue. C/D layout: VGPR r -> (M = fragBase + 8*hi + r, N = nFrag + l).
#pragma unroll
  for (int ni = 0; ni < 2; ++ni) {
    const int col = nBase + ni * 16 + l;
    const float bv = bias ? bias[col] : 0.f;
#pragma unroll
    for (int mi = 0; mi < 4; ++mi) {
      const int rowTop = mBase + mi * 16 + hi * 8;
#pragma unroll
      for (int r = 0; r < 8; ++r) {
        float v = acc[mi][ni][r] + bv;
        if (relu) v = fmaxf(v, 0.f);
        const size_t o = (size_t)(rowTop + r) * N + col;
        if (outF) outF[o] = v;
        if (outB) outB[o] = f32_to_bf16_rne(v);
      }
    }
  }
}

// ---- argmin + gather -------------------------------------------------------
__global__ void k_argmin_gather(const float* __restrict__ dot,   // [B,512]
                                const float* __restrict__ cbn,   // [512]
                                const float* __restrict__ cb,    // [512,512] f32
                                float* __restrict__ qout) {      // [B,512]
  __shared__ float sv[256];
  __shared__ int   si[256];
  const int b = blockIdx.x;
  const int tid = threadIdx.x;
  const float* row = dot + (size_t)b * 512;

  float best = INFINITY;
  int bidx = 0x7fffffff;
  for (int j = tid; j < 512; j += 256) {
    float s = cbn[j] - 2.f * row[j];      // monotone in L2 distance
    if (s < best || (s == best && j < bidx)) { best = s; bidx = j; }
  }
  sv[tid] = best; si[tid] = bidx;
  __syncthreads();
  for (int off = 128; off > 0; off >>= 1) {
    if (tid < off) {
      if (sv[tid + off] < sv[tid] ||
          (sv[tid + off] == sv[tid] && si[tid + off] < si[tid])) {
        sv[tid] = sv[tid + off]; si[tid] = si[tid + off];
      }
    }
    __syncthreads();
  }
  const int idx = si[0];
  const float* crow = cb + (size_t)idx * 512;
  float* orow = qout + (size_t)b * 512;
  for (int c = tid; c < 512; c += 256) orow[c] = crow[c];
}

// ---------------------------------------------------------------------------
extern "C" void kernel_launch(void* const* d_in, const int* in_sizes, int n_in,
                              void* d_out, int out_size, void* d_ws, size_t ws_size,
                              hipStream_t stream) {
  const float* skills = (const float*)d_in[0];
  const float* lang   = (const float*)d_in[1];
  const float* W1 = (const float*)d_in[2];
  const float* b1 = (const float*)d_in[3];
  const float* W2 = (const float*)d_in[4];
  const float* b2 = (const float*)d_in[5];
  const float* W3 = (const float*)d_in[6];
  const float* b3 = (const float*)d_in[7];
  const float* CB = (const float*)d_in[8];
  float* out = (float*)d_out;                 // [2048*512 unq | 2048*512 quant]

  char* ws = (char*)d_ws;
  const size_t MB = 1u << 20;
  unsigned short* XBF = (unsigned short*)(ws + 0);        // 4MB  (reused as H2)
  unsigned short* H1  = (unsigned short*)(ws + 4 * MB);   // 4MB  (reused: scores f32)
  unsigned short* W1T = (unsigned short*)(ws + 8 * MB);   // 2MB
  unsigned short* W2T = (unsigned short*)(ws + 10 * MB);  // 2MB
  unsigned short* W3T = (unsigned short*)(ws + 12 * MB);  // 1MB
  unsigned short* CBB = (unsigned short*)(ws + 13 * MB);  // 0.5MB
  unsigned short* UBF = (unsigned short*)(ws + 13 * MB + 512 * 1024);  // 2MB
  float*          CBN = (float*)(ws + 15 * MB + 512 * 1024);           // 2KB
  unsigned short* H2  = XBF;
  float*       SCORES = (float*)H1;

  // ---- fp32 -> bf16 staging ----
  k_concat_cvt<<<(2048 * 1024) / 256, 256, 0, stream>>>(skills, lang, XBF);
  k_transpose_cvt<<<(1024 * 1024) / 256, 256, 0, stream>>>(W1, W1T, 1024, 1024);
  k_transpose_cvt<<<(1024 * 1024) / 256, 256, 0, stream>>>(W2, W2T, 1024, 1024);
  k_transpose_cvt<<<(512 * 1024) / 256, 256, 0, stream>>>(W3, W3T, 1024, 512);
  k_copy_cvt<<<(512 * 512) / 256, 256, 0, stream>>>(CB, CBB, 512 * 512);
  k_cbnorm<<<512, 128, 0, stream>>>(CB, CBN);

  const dim3 blk(256);
  // h1 = relu(x @ W1 + b1)          M=2048 N=1024 K=1024
  k_gemm_bf16<<<dim3(1024 / 128, 2048 / 128), blk, 0, stream>>>(
      XBF, W1T, b1, nullptr, H1, 2048, 1024, 1024, 1);
  // h2 = relu(h1 @ W2 + b2)         M=2048 N=1024 K=1024
  k_gemm_bf16<<<dim3(1024 / 128, 2048 / 128), blk, 0, stream>>>(
      H1, W2T, b2, nullptr, H2, 2048, 1024, 1024, 1);
  // u = h2 @ W3 + b3  -> f32 out + bf16 copy    M=2048 N=512 K=1024
  k_gemm_bf16<<<dim3(512 / 128, 2048 / 128), blk, 0, stream>>>(
      H2, W3T, b3, out, UBF, 2048, 512, 1024, 0);
  // dot = u @ codebook^T            M=2048 N=512 K=512
  k_gemm_bf16<<<dim3(512 / 128, 2048 / 128), blk, 0, stream>>>(
      UBF, CBB, nullptr, SCORES, nullptr, 2048, 512, 512, 0);
  // nearest code + gather (fp32 codebook)
  k_argmin_gather<<<2048, 256, 0, stream>>>(SCORES, CBN, CB, out + 2048 * 512);
}
